// SliceAttention_52682068853396
// MI455X (gfx1250) — matrix operand order repaired and checked by reading, not compiled, not run on hardware
//
#include <hip/hip_runtime.h>
#include <math.h>

typedef __attribute__((ext_vector_type(16))) __bf16 v16bf;
typedef __attribute__((ext_vector_type(8)))  float  v8f;
typedef __attribute__((address_space(3))) unsigned short lds_us_t;

#define Bsz  4
#define Nseq 8192
#define Hdim 1024
#define NHd  16
#define TOK  (Bsz*Nseq)   // 32768
#define KVN  (2*Hdim)     // 2048

__device__ __forceinline__ unsigned short f2bf(float f){
  unsigned int u = __float_as_uint(f);
  u += 0x7FFFu + ((u >> 16) & 1u);
  return (unsigned short)(u >> 16);
}
__device__ __forceinline__ float bf2f(unsigned short h){
  return __uint_as_float(((unsigned int)h) << 16);
}

union FragBF { v16bf v; uint4 q[2]; };

// A-matrix frag (16x32 bf16): lane l holds row m=l%16, K = kb..kb+7 and kb+16..kb+23, kb=(l/16)*8
__device__ __forceinline__ v16bf lda(const unsigned short* s, int stride, int mbase, int k0){
  int l = (int)(threadIdx.x & 31u);
  int m = mbase + (l & 15);
  int kb = k0 + ((l >> 4) << 3);
  FragBF f;
  f.q[0] = *(const uint4*)(s + m*stride + kb);
  f.q[1] = *(const uint4*)(s + m*stride + kb + 16);
  return f.v;
}
// B-matrix frag (32x16 bf16): lane l holds col n=l%16, K = kb..kb+7 and kb+16..kb+23, kb=(l/16)*8 (K contiguous in LDS row)
__device__ __forceinline__ v16bf ldb(const unsigned short* s, int stride, int nbase, int k0){
  int l = (int)(threadIdx.x & 31u);
  int n = nbase + (l & 15);
  int kb = k0 + ((l >> 4) << 3);
  FragBF f;
  f.q[0] = *(const uint4*)(s + n*stride + kb);
  f.q[1] = *(const uint4*)(s + n*stride + kb + 16);
  return f.v;
}
__device__ __forceinline__ v8f wmma_bf16(v16bf a, v16bf b, v8f c){
  return __builtin_amdgcn_wmma_f32_16x16x32_bf16(false, a, false, b, (short)0, c, false, false);
}
#define V8F_ZERO {0.f,0.f,0.f,0.f,0.f,0.f,0.f,0.f}

// Async 16B global -> LDS copy (per lane), tracked by ASYNCcnt (ISA 08_async_tensor).
__device__ __forceinline__ void async_copy16(const unsigned short* g, unsigned short* l){
  unsigned int la = (unsigned int)(unsigned long long)(lds_us_t*)l;
  asm volatile("global_load_async_to_lds_b128 %0, %1, off"
               :: "v"(la), "v"(g) : "memory");
}
__device__ __forceinline__ void wait_async_le3(){
  asm volatile("s_wait_asynccnt 0x3" ::: "memory");
}
__device__ __forceinline__ void wait_async_0(){
  asm volatile("s_wait_asynccnt 0x0" ::: "memory");
}

// ---------------- bf16 WMMA GEMM with double-buffered async LDS staging ----------------
// Out[M,N] = A[M,K](bf16) @ BT[N,K]^T(bf16, pre-transposed) + bias
template<bool OUT_BF16>
__global__ __launch_bounds__(256) void gemm_async_kernel(
    const unsigned short* __restrict__ A, const unsigned short* __restrict__ BT,
    const float* __restrict__ bias, void* __restrict__ Outp,
    int M, int N, int K)
{
  __shared__ __align__(16) unsigned short As[2][64*32];    // [m][k]
  __shared__ __align__(16) unsigned short Bs[2][128*32];   // [n][k]
  const int tid = threadIdx.x;
  const int l = tid & 31;
  const int w = tid >> 5;
  const int m0 = blockIdx.y * 64;
  const int n0 = blockIdx.x * 128;
  const int wm = (w >> 2) * 32;
  const int wn = (w & 3) * 32;
  v8f acc[2][2] = {{V8F_ZERO, V8F_ZERO}, {V8F_ZERO, V8F_ZERO}};

  const int cr = tid >> 2;          // row within tile (0..63)
  const int cc = (tid & 3) * 8;     // 8-elem (16B) column chunk

  const unsigned short* Arow  = A  + (size_t)(m0 + cr) * K + cc;
  const unsigned short* Brow0 = BT + (size_t)(n0 + cr) * K + cc;
  const unsigned short* Brow1 = BT + (size_t)(n0 + 64 + cr) * K + cc;

  // stage tile k0 into buffer buf: 3 async b128 per thread (1 A row-chunk + 2 B row-chunks)
  #define STAGE(k0, buf)                                              \
    do {                                                              \
      async_copy16(Arow  + (k0), &As[buf][cr*32 + cc]);               \
      async_copy16(Brow0 + (k0), &Bs[buf][cr*32 + cc]);               \
      async_copy16(Brow1 + (k0), &Bs[buf][(64 + cr)*32 + cc]);        \
    } while (0)

  STAGE(0, 0);
  const int KT = K >> 5;
  for (int kt = 0; kt < KT; ++kt) {
    const int buf = kt & 1;
    if (kt + 1 < KT) {
      STAGE((kt + 1) << 5, buf ^ 1);
      wait_async_le3();          // current buffer's 3 loads (older, in-order) complete
    } else {
      wait_async_0();
    }
    __syncthreads();
    v16bf a0 = lda(As[buf], 32, wm, 0);
    v16bf a1 = lda(As[buf], 32, wm + 16, 0);
    v16bf b0 = ldb(Bs[buf], 32, wn, 0);
    v16bf b1 = ldb(Bs[buf], 32, wn + 16, 0);
    acc[0][0] = wmma_bf16(a0, b0, acc[0][0]);
    acc[0][1] = wmma_bf16(a0, b1, acc[0][1]);
    acc[1][0] = wmma_bf16(a1, b0, acc[1][0]);
    acc[1][1] = wmma_bf16(a1, b1, acc[1][1]);
    __syncthreads();
  }
  #undef STAGE
  #pragma unroll
  for (int si = 0; si < 2; ++si)
  #pragma unroll
  for (int sj = 0; sj < 2; ++sj)
  #pragma unroll
  for (int r = 0; r < 8; ++r) {
    int row = m0 + wm + si * 16 + r + ((l >> 4) << 3);
    int col = n0 + wn + sj * 16 + (l & 15);
    float v = acc[si][sj][r] + bias[col];
    if (OUT_BF16) ((unsigned short*)Outp)[(size_t)row * N + col] = f2bf(v);
    else          ((float*)Outp)[(size_t)row * N + col] = v;
  }
}

// ---------------- prep: fp32 -> bf16 elementwise ----------------
__global__ void conv_f2b_kernel(const float* __restrict__ src,
                                unsigned short* __restrict__ dst, int n)
{
  int i = blockIdx.x * 256 + threadIdx.x;
  if (i < n) dst[i] = f2bf(src[i]);
}

// ---------------- prep: src[K][N] fp32 -> dstT[N][K] bf16 (LDS-tiled transpose) ----------------
__global__ __launch_bounds__(256) void transpose_conv_kernel(
    const float* __restrict__ src, unsigned short* __restrict__ dstT, int K, int N)
{
  __shared__ unsigned short t[64][65];
  const int k0 = blockIdx.y * 64;
  const int n0 = blockIdx.x * 64;
  const int tid = threadIdx.x;
  #pragma unroll
  for (int e = 0; e < 16; ++e) {
    int i = e * 256 + tid;
    int r = i >> 6, c = i & 63;
    t[r][c] = f2bf(src[(size_t)(k0 + r) * N + n0 + c]);
  }
  __syncthreads();
  #pragma unroll
  for (int e = 0; e < 16; ++e) {
    int i = e * 256 + tid;
    int r = i >> 6, c = i & 63;   // r = n-local, c = k-local
    dstT[(size_t)(n0 + r) * K + k0 + c] = t[c][r];
  }
}

// ---------------- per-token learned temperature: 0.5 + softplus(x @ w_temp + b_temp) ----------------
__global__ __launch_bounds__(256) void temp_kernel(
    const float* __restrict__ x, const float* __restrict__ w_temp,
    const float* __restrict__ b_temp, float* __restrict__ temp)
{
  int idx = blockIdx.x * 256 + threadIdx.x;   // idx = t*16 + h
  if (idx >= TOK * NHd) return;
  int t = idx >> 4, h = idx & 15;
  float s = b_temp[h];
  const float* xr = x + (size_t)t * Hdim;
  for (int k = 0; k < Hdim; ++k) s += xr[k] * w_temp[k * NHd + h];
  float sp = (s > 20.f) ? s : log1pf(__expf(s));
  temp[idx] = 0.5f + sp;
}

// ---------------- scores + softmax(m) + slice_norm/slice_token partials ----------------
// grid (Nseq/128, B*NH), block 256
__global__ __launch_bounds__(256) void slice_kernel(
    const unsigned short* __restrict__ kv, const unsigned short* __restrict__ xqb,
    const float* __restrict__ temp, unsigned short* __restrict__ W,
    float* __restrict__ snorm, float* __restrict__ stok)
{
  __shared__ __align__(16) unsigned char smem[8192 + 16384 + 32768];  // 56KB
  unsigned short* QS = (unsigned short*)smem;              // [64 m][64 d]
  unsigned short* KW = (unsigned short*)(smem + 8192);     // phase1: xk [128 n][64 d]; phase2: weights [64 m][128 n]
  float*          SS = (float*)(smem + 24576);             // scores [64 m][128 n]
  unsigned short* VT = (unsigned short*)(smem + 24576);    // phase3: xv^T [64 d][128 n] (aliases SS)

  const int tid = threadIdx.x;
  const int l = tid & 31;
  const int w = tid >> 5;
  const int n0 = blockIdx.x * 128;
  const int bh = blockIdx.y;
  const int b = bh >> 4, h = bh & 15;

  #pragma unroll
  for (int e = 0; e < 16; ++e) { int i = e * 256 + tid; QS[i] = xqb[h * 4096 + i]; }
  #pragma unroll
  for (int e = 0; e < 32; ++e) {
    int i = e * 256 + tid;
    int n = i >> 6, d = i & 63;
    KW[i] = kv[(size_t)(b * Nseq + n0 + n) * KVN + h * 64 + d];
  }
  __syncthreads();

  // scores S[m, n] = xq[m, :] . xk[n, :], wave w covers columns [w*16, w*16+16)
  {
    v8f acc[4] = {V8F_ZERO, V8F_ZERO, V8F_ZERO, V8F_ZERO};
    v16bf bk0 = ldb(KW, 64, w * 16, 0);
    v16bf bk1 = ldb(KW, 64, w * 16, 32);
    #pragma unroll
    for (int si = 0; si < 4; ++si) {
      v16bf a0 = lda(QS, 64, si * 16, 0);
      v16bf a1 = lda(QS, 64, si * 16, 32);
      acc[si] = wmma_bf16(a0, bk0, acc[si]);
      acc[si] = wmma_bf16(a1, bk1, acc[si]);
    }
    #pragma unroll
    for (int si = 0; si < 4; ++si)
    #pragma unroll
    for (int r = 0; r < 8; ++r)
      SS[(si * 16 + r + ((l >> 4) << 3)) * 128 + w * 16 + (l & 15)] = acc[si][r];
  }
  __syncthreads();

  // softmax over m per column (divided by temperature), write weights (bf16) to LDS + global
  if (tid < 128) {
    int n = tid;
    float invt = 1.0f / temp[(size_t)(b * Nseq + n0 + n) * NHd + h];
    float z[64];
    float mx = -1e30f;
    #pragma unroll
    for (int m = 0; m < 64; ++m) { z[m] = SS[m * 128 + n] * invt; mx = fmaxf(mx, z[m]); }
    float s = 0.f;
    #pragma unroll
    for (int m = 0; m < 64; ++m) { z[m] = __expf(z[m] - mx); s += z[m]; }
    float inv = 1.0f / s;
    #pragma unroll
    for (int m = 0; m < 64; ++m) {
      unsigned short pb = f2bf(z[m] * inv);
      KW[m * 128 + n] = pb;                                  // weights [m][n]
      W[((size_t)bh * 64 + m) * Nseq + n0 + n] = pb;
    }
  }
  __syncthreads();

  // load xv^T into VT (aliases SS) while other threads reduce slice_norm partials
  if (tid < 128) {
    int n = tid;
    const unsigned short* src = kv + (size_t)(b * Nseq + n0 + n) * KVN + Hdim + h * 64;
    #pragma unroll
    for (int e = 0; e < 64; ++e) VT[e * 128 + n] = src[e];
  } else if (tid >= 192) {
    int m = tid - 192;
    float s = 0.f;
    for (int n = 0; n < 128; ++n) s += bf2f(KW[m * 128 + n]);
    atomicAdd(&snorm[bh * 64 + m], s);
  }
  __syncthreads();

  // slice_token partial: ST[m,d] += sum_n weights[m,n] * xv[n,d]; 16 tiles over 8 waves
  #pragma unroll
  for (int tt = 0; tt < 2; ++tt) {
    int t = w * 2 + tt;
    int si = t >> 2, sj = t & 3;
    v8f acc = V8F_ZERO;
    #pragma unroll
    for (int ks = 0; ks < 4; ++ks) {
      v16bf a  = lda(KW, 128, si * 16, ks * 32);
      v16bf bb = ldb(VT, 128, sj * 16, ks * 32);
      acc = wmma_bf16(a, bb, acc);
    }
    #pragma unroll
    for (int r = 0; r < 8; ++r)
      atomicAdd(&stok[(size_t)bh * 4096 + (si * 16 + r + ((l >> 4) << 3)) * 64 + sj * 16 + (l & 15)],
                acc[r]);
  }
}

// ---------------- tiny intra-slice "attention" (elementwise, softmax over d) ----------------
// grid (B*NH), block 64 (thread = m)
__global__ __launch_bounds__(64) void intra_kernel(
    const float* __restrict__ stok, const float* __restrict__ snorm,
    const float* __restrict__ qkv_proj, unsigned short* __restrict__ ostT)
{
  int bh = blockIdx.x;
  int h = bh & 15;
  int m = threadIdx.x;
  float inv = 1.0f / (snorm[bh * 64 + m] + 1e-5f);
  float st[64];
  for (int d = 0; d < 64; ++d) st[d] = stok[(size_t)bh * 4096 + m * 64 + d] * inv;
  const float* P = qkv_proj + (size_t)h * 64 * 192;
  float dots[64], vv[64];
  for (int e = 0; e < 64; ++e) {
    float q = 0.f, k = 0.f, v = 0.f;
    for (int d = 0; d < 64; ++d) {
      float s = st[d];
      const float* pr = P + d * 192;
      q += s * pr[e];
      k += s * pr[64 + e];
      v += s * pr[128 + e];
    }
    dots[e] = q * k * 0.125f;   // SCALE = 64^-0.5
    vv[e] = v;
  }
  float mx = -1e30f;
  for (int e = 0; e < 64; ++e) mx = fmaxf(mx, dots[e]);
  float s = 0.f;
  for (int e = 0; e < 64; ++e) { dots[e] = __expf(dots[e] - mx); s += dots[e]; }
  float invs = 1.0f / s;
  // store transposed [d][m] (bf16) so scatter's A-matrix has K=m contiguous
  for (int e = 0; e < 64; ++e)
    ostT[(size_t)bh * 4096 + e * 64 + m] = f2bf(dots[e] * invs * vv[e]);
}

// ---------------- scatter back: out_x[n, h*64+d] = sum_m ost[m,d] * W[m,n] ----------------
// grid (Nseq/256, B*NH), block 256
__global__ __launch_bounds__(256) void scatter_kernel(
    const unsigned short* __restrict__ W, const unsigned short* __restrict__ ostT,
    unsigned short* __restrict__ outx)
{
  __shared__ __align__(16) unsigned short AT[64 * 64];    // ost^T [d][m]
  __shared__ __align__(16) unsigned short BW[256 * 64];   // W^T   [n][m]
  const int tid = threadIdx.x;
  const int l = tid & 31;
  const int w = tid >> 5;
  const int n0 = blockIdx.x * 256;
  const int bh = blockIdx.y;
  const int b = bh >> 4, h = bh & 15;
  #pragma unroll
  for (int e = 0; e < 16; ++e) { int i = e * 256 + tid; AT[i] = ostT[(size_t)bh * 4096 + i]; }
  {
    int n = tid;
    #pragma unroll
    for (int e = 0; e < 64; ++e)
      BW[n * 64 + e] = W[((size_t)bh * 64 + e) * Nseq + n0 + n];
  }
  __syncthreads();
  #pragma unroll
  for (int si = 0; si < 4; ++si)
  #pragma unroll
  for (int sj = 0; sj < 2; ++sj) {
    v8f acc = V8F_ZERO;
    v16bf a0 = lda(AT, 64, si * 16, 0);
    v16bf a1 = lda(AT, 64, si * 16, 32);
    v16bf b0 = ldb(BW, 64, w * 32 + sj * 16, 0);
    v16bf b1 = ldb(BW, 64, w * 32 + sj * 16, 32);
    acc = wmma_bf16(a0, b0, acc);
    acc = wmma_bf16(a1, b1, acc);
    #pragma unroll
    for (int r = 0; r < 8; ++r) {
      int d = si * 16 + r + ((l >> 4) << 3);
      int n = w * 32 + sj * 16 + (l & 15);
      outx[(size_t)(b * Nseq + n0 + n) * Hdim + h * 64 + d] = f2bf(acc[r]);
    }
  }
}

extern "C" void kernel_launch(void* const* d_in, const int* in_sizes, int n_in,
                              void* d_out, int out_size, void* d_ws, size_t ws_size,
                              hipStream_t stream)
{
  (void)in_sizes; (void)n_in; (void)out_size; (void)ws_size;
  const float* x        = (const float*)d_in[0];
  // d_in[1] = 'c' is unused by the reference
  const float* w_kv     = (const float*)d_in[2];
  const float* b_kv     = (const float*)d_in[3];
  const float* w_temp   = (const float*)d_in[4];
  const float* b_temp   = (const float*)d_in[5];
  const float* xq       = (const float*)d_in[6];
  const float* qkv_proj = (const float*)d_in[7];
  const float* w_out    = (const float*)d_in[8];
  const float* b_out    = (const float*)d_in[9];
  float* out = (float*)d_out;

  size_t off = 0;
  char* wsb = (char*)d_ws;
  auto alloc = [&](size_t bytes) -> void* {
    void* p = wsb + off;
    off += (bytes + 255) & ~(size_t)255;
    return p;
  };
  unsigned short* kv    = (unsigned short*)alloc((size_t)TOK * KVN * 2);       // 128 MB
  unsigned short* Wt    = (unsigned short*)alloc((size_t)64 * 64 * Nseq * 2);  // 64 MB
  unsigned short* outx  = (unsigned short*)alloc((size_t)TOK * Hdim * 2);      // 64 MB
  unsigned short* xbf   = (unsigned short*)alloc((size_t)TOK * Hdim * 2);      // 64 MB
  unsigned short* wkvT  = (unsigned short*)alloc((size_t)KVN * Hdim * 2);      // 4 MB
  unsigned short* woutT = (unsigned short*)alloc((size_t)Hdim * Hdim * 2);     // 2 MB
  float*          temp  = (float*)alloc((size_t)TOK * NHd * 4);                // 2 MB
  unsigned short* xqb   = (unsigned short*)alloc((size_t)NHd * 64 * 64 * 2);
  float*          snorm = (float*)alloc((size_t)64 * 64 * 4);
  float*          stok  = (float*)alloc((size_t)64 * 64 * 64 * 4);
  unsigned short* ostT  = (unsigned short*)alloc((size_t)64 * 64 * 64 * 2);

  hipMemsetAsync(snorm, 0, (size_t)64 * 64 * 4, stream);
  hipMemsetAsync(stok, 0, (size_t)64 * 64 * 64 * 4, stream);

  // prep: bf16 casts + weight transposes
  conv_f2b_kernel<<<(TOK * Hdim + 255) / 256, 256, 0, stream>>>(x, xbf, TOK * Hdim);
  conv_f2b_kernel<<<(NHd * 64 * 64 + 255) / 256, 256, 0, stream>>>(xq, xqb, NHd * 64 * 64);
  {
    dim3 g(KVN / 64, Hdim / 64);
    transpose_conv_kernel<<<g, 256, 0, stream>>>(w_kv, wkvT, Hdim, KVN);
  }
  {
    dim3 g(Hdim / 64, Hdim / 64);
    transpose_conv_kernel<<<g, 256, 0, stream>>>(w_out, woutT, Hdim, Hdim);
  }
  temp_kernel<<<(TOK * NHd + 255) / 256, 256, 0, stream>>>(x, w_temp, b_temp, temp);

  // kv = x @ w_kv + b_kv  (bf16 async-staged GEMM, bf16 out)
  {
    dim3 g(KVN / 128, TOK / 64);
    gemm_async_kernel<true><<<g, 256, 0, stream>>>(xbf, wkvT, b_kv, kv, TOK, KVN, Hdim);
  }
  // scores -> softmax over m -> weights + slice_norm/slice_token partials
  {
    dim3 g(Nseq / 128, Bsz * NHd);
    slice_kernel<<<g, 256, 0, stream>>>(kv, xqb, temp, Wt, snorm, stok);
  }
  intra_kernel<<<Bsz * NHd, 64, 0, stream>>>(stok, snorm, qkv_proj, ostT);
  // out_x = W^T @ ost (per head), token-major bf16
  {
    dim3 g(Nseq / 256, Bsz * NHd);
    scatter_kernel<<<g, 256, 0, stream>>>(Wt, ostT, outx);
  }
  // out = out_x @ w_out + b_out  (bf16 async-staged GEMM, fp32 out)
  {
    dim3 g(Hdim / 128, TOK / 64);
    gemm_async_kernel<false><<<g, 256, 0, stream>>>(outx, woutT, b_out, out, TOK, Hdim, Hdim);
  }
}
